// EncoderDecoderAttention_10136122818720
// MI455X (gfx1250) — matrix operand order repaired and checked by reading, not compiled, not run on hardware
//
#include <hip/hip_runtime.h>

// ---------------------------------------------------------------------------
// Problem constants (from the reference)
// ---------------------------------------------------------------------------
#define D_MODEL 512
#define NHEADS  8
#define DK      64
#define DV      64
#define SQ      2048
#define SKV     2048

// ---------------------------------------------------------------------------
// CDNA5 WMMA types
// ---------------------------------------------------------------------------
typedef __attribute__((ext_vector_type(16))) __bf16 v16bf;
typedef __attribute__((ext_vector_type(8)))  float  v8f;

struct Bf16x16 { uint4 lo, hi; };   // 32 bytes == one v16bf fragment

__device__ __forceinline__ v8f wmma_bf16(v16bf a, v16bf b, v8f c) {
  // D = A(16x32 bf16) * B(32x16 bf16) + C(16x16 f32)
  return __builtin_amdgcn_wmma_f32_16x16x32_bf16(
      /*neg_a=*/false, a, /*neg_b=*/false, b,
      /*c_mod=*/(short)0, c, /*reuse_a=*/false, /*reuse_b=*/false);
}

// f32 -> bf16 (round to nearest even)
__device__ __forceinline__ unsigned short f2bf(float f) {
  unsigned u = __builtin_bit_cast(unsigned, f);
  unsigned r = u + 0x7fffu + ((u >> 16) & 1u);
  return (unsigned short)(r >> 16);
}

// A-fragment: row-per-lane. Loads elements [koff..koff+7] and [koff+16..koff+23]
// of `row` (bf16). koff must already include the (lane<16 ? 0 : 8) half-select.
template <typename PT>
__device__ __forceinline__ v16bf load_a_frag(const PT* row, int koff) {
  Bf16x16 t;
  t.lo = *reinterpret_cast<const uint4*>(row + koff);
  t.hi = *reinterpret_cast<const uint4*>(row + koff + 16);
  return __builtin_bit_cast(v16bf, t);
}

// B-fragment: column-per-lane, elements [col..col+7] and [col+16..col+23]
// (caller already applied the (lane<16 ? 0 : 8) K-half select).
template <typename PT>
__device__ __forceinline__ v16bf load_b_frag(const PT* col) {
  Bf16x16 t;
  t.lo = *reinterpret_cast<const uint4*>(col);
  t.hi = *reinterpret_cast<const uint4*>(col + 16);
  return __builtin_bit_cast(v16bf, t);
}

// ---------------------------------------------------------------------------
// f32 -> bf16 conversion kernels
// ---------------------------------------------------------------------------
__global__ void cvt_bf16_kernel(const float* __restrict__ in,
                                unsigned short* __restrict__ out, int n) {
  int i = blockIdx.x * blockDim.x + threadIdx.x;
  if (i < n) out[i] = f2bf(in[i]);
}

// in: [batch, rows, cols] f32  ->  out: [batch, cols, rows] bf16
__global__ void cvt_bf16_tr_kernel(const float* __restrict__ in,
                                   unsigned short* __restrict__ out,
                                   int rows, int cols, int total) {
  int i = blockIdx.x * blockDim.x + threadIdx.x;
  if (i >= total) return;
  int rc = rows * cols;
  int b = i / rc;
  int r = (i / cols) % rows;
  int c = i % cols;
  out[(b * cols + c) * rows + r] = f2bf(in[i]);
}

// ---------------------------------------------------------------------------
// Per-head projection GEMM (software-pipelined):
//   out[h, s, 0:64] = X[s, :] (bf16, [S,512]) * W[h]^T  + bias[h]
// WT layout: [H, 64, 512] bf16 (column n of W[h] is a contiguous 512-row).
// transposeOut == 0 -> out[h, s, n]   (row-major, q/k)
// transposeOut == 1 -> out[h, n, s]   (V stored transposed for PV WMMA)
// 128 threads = 4 waves; each wave computes one 16x64 tile.
// grid = (S/16/4, H)
// ---------------------------------------------------------------------------
__global__ void proj_gemm_kernel(const unsigned short* __restrict__ X,
                                 const unsigned short* __restrict__ WT,
                                 const float* __restrict__ bias,
                                 unsigned short* __restrict__ out,
                                 int S, int transposeOut) {
  const int lane = threadIdx.x & 31;
  const int wave = threadIdx.x >> 5;
  const int s0   = (blockIdx.x * 4 + wave) * 16;
  const int h    = blockIdx.y;

  const unsigned short* xrow  = X + (size_t)(s0 + (lane & 15)) * D_MODEL;
  const unsigned short* wbase = WT + (size_t)h * 64 * D_MODEL;
  const int ahalf = (lane < 16) ? 0 : 8;
  const int bhalf = (lane < 16) ? 0 : 8;
  const int ncol  = lane & 15;
  const int KS    = D_MODEL / 32;   // 16 k-steps

  // prologue: fragments for k-step 0
  v16bf a0 = load_a_frag(xrow, ahalf);
  v16bf b0[4];
#pragma unroll
  for (int f = 0; f < 4; ++f)
    b0[f] = load_b_frag(wbase + (size_t)(f * 16 + ncol) * D_MODEL + bhalf);

  v8f acc[4] = {};
  for (int ks = 0; ks < KS - 1; ++ks) {
    // issue next k-step's loads before consuming the current fragments
    v16bf a1 = load_a_frag(xrow, (ks + 1) * 32 + ahalf);
    v16bf b1[4];
#pragma unroll
    for (int f = 0; f < 4; ++f)
      b1[f] = load_b_frag(wbase + (size_t)(f * 16 + ncol) * D_MODEL +
                          (ks + 1) * 32 + bhalf);
#pragma unroll
    for (int f = 0; f < 4; ++f) acc[f] = wmma_bf16(a0, b0[f], acc[f]);
    a0 = a1;
#pragma unroll
    for (int f = 0; f < 4; ++f) b0[f] = b1[f];
  }
#pragma unroll
  for (int f = 0; f < 4; ++f) acc[f] = wmma_bf16(a0, b0[f], acc[f]);

  const int rr = (lane < 16) ? 0 : 8;
#pragma unroll
  for (int f = 0; f < 4; ++f) {
    int n = f * 16 + ncol;
    float bv = bias[h * 64 + n];
#pragma unroll
    for (int g = 0; g < 8; ++g) {
      int m = s0 + g + rr;
      unsigned short bf = f2bf(acc[f][g] + bv);
      if (!transposeOut) out[((size_t)h * S + m) * 64 + n] = bf;
      else               out[((size_t)h * 64 + n) * S + m] = bf;
    }
  }
}

// ---------------------------------------------------------------------------
// Flash-style attention. One wave per (head, 16-query tile).
//   qB: [H, SQ, 64] bf16      kB: [H, SKV, 64] bf16     vT: [H, 64, SKV] bf16
//   concat: [SQ, H*DV] bf16   (head h occupies columns h*64..h*64+63)
// Online softmax over 32-wide KV chunks. K fragments are double-buffered
// across chunks and V fragments are loaded before the softmax so that
// global-load latency overlaps the serial softmax / LDS work.
// grid = (SQ/16, H), block = 32
// ---------------------------------------------------------------------------
__global__ void attn_kernel(const unsigned short* __restrict__ qB,
                            const unsigned short* __restrict__ kB,
                            const unsigned short* __restrict__ vT,
                            unsigned short* __restrict__ concat) {
  __shared__ __align__(16) float          lds_s[16 * 32];
  __shared__ __align__(16) unsigned short lds_p[16 * 32];
  __shared__ __align__(16) float          lds_f[16];

  const int lane  = threadIdx.x & 31;
  const int s0    = blockIdx.x * 16;
  const int h     = blockIdx.y;
  const int ahalf = (lane < 16) ? 0 : 8;
  const int bhalf = (lane < 16) ? 0 : 8;
  const int rr    = (lane < 16) ? 0 : 8;
  const int ncol  = lane & 15;
  const float scale = 0.125f;  // 1/sqrt(64)

  // Q fragments for the whole tile (K = 64 -> two 16x32 A-fragments)
  const unsigned short* qrow = qB + ((size_t)h * SQ + s0 + ncol) * DK;
  v16bf aq0 = load_a_frag(qrow, 0  + ahalf);
  v16bf aq1 = load_a_frag(qrow, 32 + ahalf);

  const unsigned short* kbase = kB + (size_t)h * SKV * DK;
  const unsigned short* vbase = vT + (size_t)h * DV * SKV;

  // prologue: K fragments for the first KV chunk  [n-tile][k-step]
  v16bf kb[2][2];
#pragma unroll
  for (int n = 0; n < 2; ++n) {
    const unsigned short* krow = kbase + (size_t)(n * 16 + ncol) * DK;
    kb[n][0] = load_b_frag(krow + bhalf);
    kb[n][1] = load_b_frag(krow + 32 + bhalf);
  }

  v8f   o[4] = {};
  float mrow = -3.0e38f;
  float lrow = 0.0f;

  for (int t0 = 0; t0 < SKV; t0 += 32) {
    // ---- S = (Q K^T) * scale  for a 16x32 chunk ----
    v8f s[2] = {};
#pragma unroll
    for (int n = 0; n < 2; ++n) {
      s[n] = wmma_bf16(aq0, kb[n][0], s[n]);
      s[n] = wmma_bf16(aq1, kb[n][1], s[n]);
    }

    // issue next chunk's K fragments now; they'll land during the softmax
    if (t0 + 32 < SKV) {
      __builtin_prefetch(kbase + (size_t)(t0 + 64 + ncol) * DK, 0, 1);
#pragma unroll
      for (int n = 0; n < 2; ++n) {
        const unsigned short* krow = kbase + (size_t)(t0 + 32 + n * 16 + ncol) * DK;
        kb[n][0] = load_b_frag(krow + bhalf);
        kb[n][1] = load_b_frag(krow + 32 + bhalf);
      }
    }

    // V fragments for *this* chunk, also issued before the softmax
    v16bf bv[4];
#pragma unroll
    for (int f = 0; f < 4; ++f)
      bv[f] = load_b_frag(vbase + (size_t)(f * 16 + ncol) * SKV + t0 + bhalf);

#pragma unroll
    for (int n = 0; n < 2; ++n)
#pragma unroll
      for (int g = 0; g < 8; ++g)
        lds_s[(g + rr) * 32 + n * 16 + ncol] = s[n][g] * scale;
    __syncthreads();

    // ---- online softmax update (one row per lane 0..15) ----
    if (lane < 16) {
      const float* srow = lds_s + lane * 32;
      float tmax = srow[0];
#pragma unroll
      for (int j = 1; j < 32; ++j) tmax = fmaxf(tmax, srow[j]);
      float mnew   = fmaxf(mrow, tmax);
      float factor = __expf(mrow - mnew);
      float sum = 0.0f;
#pragma unroll
      for (int j = 0; j < 32; ++j) {
        float p = __expf(srow[j] - mnew);
        sum += p;
        lds_p[lane * 32 + j] = f2bf(p);
      }
      lrow = lrow * factor + sum;
      mrow = mnew;
      lds_f[lane] = factor;
    }
    __syncthreads();

    // ---- rescale running O, then O += P * V ----
    float fac[8];
#pragma unroll
    for (int g = 0; g < 8; ++g) fac[g] = lds_f[g + rr];
#pragma unroll
    for (int f = 0; f < 4; ++f)
#pragma unroll
      for (int g = 0; g < 8; ++g) o[f][g] *= fac[g];

    v16bf ap = load_a_frag(lds_p + ncol * 32, ahalf);
#pragma unroll
    for (int f = 0; f < 4; ++f) o[f] = wmma_bf16(ap, bv[f], o[f]);
    __syncthreads();  // protect lds_s/lds_p for the next chunk
  }

  // ---- final 1/l normalization and store into concat buffer ----
  if (lane < 16) lds_f[lane] = 1.0f / lrow;
  __syncthreads();
  float inv[8];
#pragma unroll
  for (int g = 0; g < 8; ++g) inv[g] = lds_f[g + rr];
#pragma unroll
  for (int f = 0; f < 4; ++f) {
    int n = h * 64 + f * 16 + ncol;
#pragma unroll
    for (int g = 0; g < 8; ++g) {
      int m = s0 + g + rr;
      concat[(size_t)m * (NHEADS * DV) + n] = f2bf(o[f][g] * inv[g]);
    }
  }
}

// ---------------------------------------------------------------------------
// Final projection (software-pipelined):
//   out[2048,512] = concat(bf16) @ Wo + bo   (f32 output)
// WoT layout: [512, 512] bf16 with WoT[n][d] = Wo[d][n].
// 128 threads = 4 waves; each wave one 16x64 tile. grid = (SQ/16/4, 512/64)
// ---------------------------------------------------------------------------
__global__ void out_gemm_kernel(const unsigned short* __restrict__ X,
                                const unsigned short* __restrict__ WoT,
                                const float* __restrict__ bo,
                                float* __restrict__ out) {
  const int lane = threadIdx.x & 31;
  const int wave = threadIdx.x >> 5;
  const int s0   = (blockIdx.x * 4 + wave) * 16;
  const int n0   = blockIdx.y * 64;
  const int ahalf = (lane < 16) ? 0 : 8;
  const int bhalf = (lane < 16) ? 0 : 8;
  const int ncol  = lane & 15;
  const int KS    = D_MODEL / 32;

  const unsigned short* xrow = X + (size_t)(s0 + ncol) * D_MODEL;

  v16bf a0 = load_a_frag(xrow, ahalf);
  v16bf b0[4];
#pragma unroll
  for (int f = 0; f < 4; ++f)
    b0[f] = load_b_frag(WoT + (size_t)(n0 + f * 16 + ncol) * D_MODEL + bhalf);

  v8f acc[4] = {};
  for (int ks = 0; ks < KS - 1; ++ks) {
    v16bf a1 = load_a_frag(xrow, (ks + 1) * 32 + ahalf);
    v16bf b1[4];
#pragma unroll
    for (int f = 0; f < 4; ++f)
      b1[f] = load_b_frag(WoT + (size_t)(n0 + f * 16 + ncol) * D_MODEL +
                          (ks + 1) * 32 + bhalf);
#pragma unroll
    for (int f = 0; f < 4; ++f) acc[f] = wmma_bf16(a0, b0[f], acc[f]);
    a0 = a1;
#pragma unroll
    for (int f = 0; f < 4; ++f) b0[f] = b1[f];
  }
#pragma unroll
  for (int f = 0; f < 4; ++f) acc[f] = wmma_bf16(a0, b0[f], acc[f]);

  const int rr = (lane < 16) ? 0 : 8;
#pragma unroll
  for (int f = 0; f < 4; ++f) {
    int n = n0 + f * 16 + ncol;
    float bv = bo[n];
#pragma unroll
    for (int g = 0; g < 8; ++g) {
      int m = s0 + g + rr;
      out[(size_t)m * D_MODEL + n] = acc[f][g] + bv;
    }
  }
}

// ---------------------------------------------------------------------------
// Launch
// ---------------------------------------------------------------------------
extern "C" void kernel_launch(void* const* d_in, const int* in_sizes, int n_in,
                              void* d_out, int out_size, void* d_ws, size_t ws_size,
                              hipStream_t stream) {
  (void)in_sizes; (void)n_in; (void)out_size; (void)ws_size;

  const float* emb = (const float*)d_in[0];   // [SQ, 512]
  const float* Kin = (const float*)d_in[1];   // [SKV, 512]
  const float* Vin = (const float*)d_in[2];   // [SKV, 512]
  const float* Wq  = (const float*)d_in[3];   // [H, 512, 64]
  const float* bq  = (const float*)d_in[4];   // [H, 64]
  const float* Wk  = (const float*)d_in[5];
  const float* bk  = (const float*)d_in[6];
  const float* Wv  = (const float*)d_in[7];
  const float* bv  = (const float*)d_in[8];
  const float* Wo  = (const float*)d_in[9];   // [512, 512]
  const float* bo  = (const float*)d_in[10];  // [512]
  float* out = (float*)d_out;

  // ---- workspace layout (bf16 stored as unsigned short) ----
  char* ws = (char*)d_ws;
  size_t off = 0;
  auto alloc = [&](size_t bytes) -> unsigned short* {
    unsigned short* p = (unsigned short*)(ws + off);
    off += (bytes + 255) & ~(size_t)255;
    return p;
  };
  unsigned short* embB = alloc((size_t)SQ * D_MODEL * 2);
  unsigned short* KB   = alloc((size_t)SKV * D_MODEL * 2);
  unsigned short* VB   = alloc((size_t)SKV * D_MODEL * 2);
  unsigned short* WqT  = alloc((size_t)NHEADS * 64 * D_MODEL * 2);
  unsigned short* WkT  = alloc((size_t)NHEADS * 64 * D_MODEL * 2);
  unsigned short* WvT  = alloc((size_t)NHEADS * 64 * D_MODEL * 2);
  unsigned short* WoT  = alloc((size_t)D_MODEL * D_MODEL * 2);
  unsigned short* qB   = alloc((size_t)NHEADS * SQ * DK * 2);
  unsigned short* kB   = alloc((size_t)NHEADS * SKV * DK * 2);
  unsigned short* vTB  = alloc((size_t)NHEADS * DV * SKV * 2);
  unsigned short* catB = alloc((size_t)SQ * NHEADS * DV * 2);

  // ---- 1) precision conversion (f32 -> bf16, weights transposed) ----
  {
    int n = SQ * D_MODEL;
    cvt_bf16_kernel<<<(n + 255) / 256, 256, 0, stream>>>(emb, embB, n);
    cvt_bf16_kernel<<<(n + 255) / 256, 256, 0, stream>>>(Kin, KB, n);
    cvt_bf16_kernel<<<(n + 255) / 256, 256, 0, stream>>>(Vin, VB, n);
    int nw = NHEADS * D_MODEL * 64;
    cvt_bf16_tr_kernel<<<(nw + 255) / 256, 256, 0, stream>>>(Wq, WqT, D_MODEL, 64, nw);
    cvt_bf16_tr_kernel<<<(nw + 255) / 256, 256, 0, stream>>>(Wk, WkT, D_MODEL, 64, nw);
    cvt_bf16_tr_kernel<<<(nw + 255) / 256, 256, 0, stream>>>(Wv, WvT, D_MODEL, 64, nw);
    int no = D_MODEL * D_MODEL;
    cvt_bf16_tr_kernel<<<(no + 255) / 256, 256, 0, stream>>>(Wo, WoT, D_MODEL, D_MODEL, no);
  }

  // ---- 2) Q/K/V projections (WMMA bf16 GEMMs, 4 waves/block) ----
  {
    dim3 grid(SQ / 16 / 4, NHEADS);
    proj_gemm_kernel<<<grid, 128, 0, stream>>>(embB, WqT, bq, qB, SQ, 0);
    proj_gemm_kernel<<<grid, 128, 0, stream>>>(KB,   WkT, bk, kB, SKV, 0);
    proj_gemm_kernel<<<grid, 128, 0, stream>>>(VB,   WvT, bv, vTB, SKV, 1);  // transposed
  }

  // ---- 3) attention (flash-style, online softmax) ----
  {
    dim3 grid(SQ / 16, NHEADS);
    attn_kernel<<<grid, 32, 0, stream>>>(qB, kB, vTB, catB);
  }

  // ---- 4) output projection ----
  {
    dim3 grid(SQ / 16 / 4, D_MODEL / 64);
    out_gemm_kernel<<<grid, 128, 0, stream>>>(catB, WoT, bo, out);
  }
}
